// CRFLoss_10196252360958
// MI455X (gfx1250) — compile-verified, matrix operand order checked by reading, not run on hardware
//
#include <hip/hip_runtime.h>

typedef __attribute__((ext_vector_type(16))) _Float16 v16h;
typedef __attribute__((ext_vector_type(8)))  float    v8f;

#define B_  64
#define T_  512
#define K_  48
#define TILE_ELEMS (K_ * K_)          // 2304 floats
#define TILE_BYTES (TILE_ELEMS * 4)   // 9216 bytes (16B aligned: 576 * 16)
#define NTHREADS 96
#define CHUNKS (TILE_BYTES / 16)      // 576 b128 chunks
#define CPT (CHUNKS / NTHREADS)       // 6 chunks per thread

// Async copy of one 48x48 f32 tile: global -> LDS, tracked by ASYNCcnt.
__device__ __forceinline__ void async_tile_load(const float* gbase,
                                                unsigned lds_base, int tid) {
  #pragma unroll
  for (int j = 0; j < CPT; ++j) {
    const int c = tid + j * NTHREADS;
    unsigned long long ga =
        (unsigned long long)(uintptr_t)gbase + (unsigned long long)c * 16ull;
    unsigned la = lds_base + (unsigned)c * 16u;
    asm volatile("global_load_async_to_lds_b128 %0, %1, off"
                 :: "v"(la), "v"(ga) : "memory");
  }
}

__device__ __forceinline__ void wait_async0() {
  asm volatile("s_wait_asynccnt 0" ::: "memory");
}

// Max over fsbuf[0..47], replicated into every lane of every wave (no LDS
// round-trip, no extra barrier).
__device__ __forceinline__ float wave_max48(const float* fsbuf, int lane) {
  float m = fmaxf(fsbuf[lane], fsbuf[lane + 16]);   // covers 0..47 (16..31 dup)
  #pragma unroll
  for (int off = 16; off; off >>= 1) m = fmaxf(m, __shfl_xor(m, off, 32));
  return m;
}

// One workgroup (3 waves) per batch element. Wave w owns columns 16w..16w+15.
// Per step: logsumexp step as linear-space column sums via two
// v_wmma_f32_16x16x32_f16 (ones(16x32) x W(32x16)), tiles double-buffered in
// LDS via global_load_async_to_lds_b128.
__global__ __launch_bounds__(NTHREADS)
void crf_forward(const float* __restrict__ scores,
                 const int*   __restrict__ targets,
                 const int*   __restrict__ lengths,
                 const int*   __restrict__ start_idx,
                 const int*   __restrict__ end_idx,
                 float*       __restrict__ partials) {
  __shared__ __align__(16) float tiles[2][TILE_ELEMS];  // 18432 B
  __shared__ float fsbuf[K_];
  __shared__ int   tgt[T_];                             // 2048 B

  const int tid   = threadIdx.x;
  const int wave  = tid >> 5;
  const int lane  = tid & 31;
  const int nloc  = lane & 15;
  const int khalf = (lane >> 4) & 1;
  const int n     = wave * 16 + nloc;        // output column 0..47
  const int b     = blockIdx.x;

  const int len  = lengths[b];               // in [1, T]
  const int sidx = start_idx[0];
  const int eidx = end_idx[0];

  const float* sb = scores + (size_t)b * T_ * TILE_ELEMS;
  const unsigned ldsb[2] = {(unsigned)(uintptr_t)&tiles[0][0],
                            (unsigned)(uintptr_t)&tiles[1][0]};

  // Stage fs0 = scores[b,0,start,:], the targets row, and tile 0.
  if (tid < K_) fsbuf[tid] = sb[(size_t)sidx * K_ + tid];
  for (int i = tid; i < T_; i += NTHREADS) tgt[i] = targets[(size_t)b * T_ + i];
  async_tile_load(sb, ldsb[0], tid);
  wait_async0();
  __syncthreads();

  float mx = wave_max48(fsbuf, lane);

  v16h ones;
  #pragma unroll
  for (int e = 0; e < 16; ++e) ones[e] = (_Float16)1.0f;

  float gold = 0.0f;

  for (int t = 0; t < len; ++t) {
    const float* st = &tiles[t & 1][0];

    // Kick off the async copy of the next tile into the other buffer.
    if (t + 1 < len)
      async_tile_load(sb + (size_t)(t + 1) * TILE_ELEMS, ldsb[(t + 1) & 1], tid);

    // W chunks in the 16-bit B-matrix (32x16) VGPR layout:
    //   lanes 0-15 : K = 0..7  (elems 0-7),  K = 16..23 (elems 8-15)
    //   lanes 16-31: K = 8..15 (elems 0-7),  K = 24..31 (elems 8-15)
    v16h w0, w1;
    #pragma unroll
    for (int e = 0; e < 16; ++e) {
      int k   = ((e >> 3) << 4) + (khalf << 3) + (e & 7);   // 0..31
      float a = st[k * K_ + n] + fsbuf[k] - mx;
      w0[e] = (_Float16)__expf(a);
    }
    #pragma unroll
    for (int e = 0; e < 8; ++e) {
      int k   = 32 + (khalf << 3) + e;                      // 32..47
      float a = st[k * K_ + n] + fsbuf[k] - mx;
      w1[e] = (_Float16)__expf(a);
    }
    #pragma unroll
    for (int e = 8; e < 16; ++e) w1[e] = (_Float16)0.0f;    // K pad 48..63

    // Column sums over K: all D rows identical => acc[0] = colsum[lane & 15].
    v8f acc = {};
    acc = __builtin_amdgcn_wmma_f32_16x16x32_f16(false, ones, false, w0,
                                                 (short)0, acc, false, false);
    acc = __builtin_amdgcn_wmma_f32_16x16x32_f16(false, ones, false, w1,
                                                 (short)0, acc, false, false);

    float newfs = mx + __logf(acc[0]);

    if (tid == 0) gold += st[tgt[t]];        // gold gather from the LDS tile

    __syncthreads();                          // old fsbuf fully consumed
    if (lane < 16) fsbuf[n] = newfs;          // one writer per column
    wait_async0();                            // next tile landed in LDS
    __syncthreads();                          // fsbuf + next tile visible

    mx = wave_max48(fsbuf, lane);             // per-wave, no extra barrier
  }

  if (tid == 0) partials[b] = fsbuf[eidx] - gold;
}

// Fixed-order reduction => deterministic across graph replays.
__global__ void crf_finalize(const float* __restrict__ partials,
                             float* __restrict__ out) {
  if (threadIdx.x == 0) {
    float s = 0.0f;
    for (int i = 0; i < B_; ++i) s += partials[i];
    out[0] = s / (float)B_;
  }
}

extern "C" void kernel_launch(void* const* d_in, const int* in_sizes, int n_in,
                              void* d_out, int out_size, void* d_ws, size_t ws_size,
                              hipStream_t stream) {
  const float* scores  = (const float*)d_in[0];
  const int*   targets = (const int*)d_in[1];
  const int*   lengths = (const int*)d_in[2];
  const int*   sidx    = (const int*)d_in[3];
  const int*   eidx    = (const int*)d_in[4];
  float* partials = (float*)d_ws;            // 64 floats of scratch

  crf_forward<<<B_, NTHREADS, 0, stream>>>(scores, targets, lengths, sidx, eidx,
                                           partials);
  crf_finalize<<<1, 32, 0, stream>>>(partials, (float*)d_out);
}
